// MultiHeadGraphAttention_39685497815574
// MI455X (gfx1250) — matrix operand order repaired and checked by reading, not compiled, not run on hardware
//
#include <hip/hip_runtime.h>
#include <hip/hip_bf16.h>

typedef __attribute__((ext_vector_type(16))) _Float16 v16h;
typedef __attribute__((ext_vector_type(8)))  _Float16 v8h;
typedef __attribute__((ext_vector_type(8)))  float    v8f;
typedef __attribute__((ext_vector_type(4)))  unsigned int v4u;
typedef __attribute__((ext_vector_type(8)))  int v8i;
typedef __attribute__((ext_vector_type(4)))  int v4i;

#define BSZ   512
#define NN    128
#define HEADS 8
#define DIN   64
#define DOUT  64

union V16U { v16h v; v8h h[2]; };

#if defined(__gfx1250__) && __has_builtin(__builtin_amdgcn_tensor_load_to_lds)
#define USE_TDM 1
#else
#define USE_TDM 0
#endif

__device__ __forceinline__ float fast_tanh(float x) {
#if defined(__gfx1250__) && __has_builtin(__builtin_amdgcn_tanhf)
    return __builtin_amdgcn_tanhf(x);        // native V_TANH_F32
#elif defined(__gfx1250__) && __has_builtin(__builtin_amdgcn_tanh_f32)
    return __builtin_amdgcn_tanh_f32(x);
#else
    return tanhf(x);
#endif
}

__global__ __launch_bounds__(256, 1)
void MultiHeadGraphAttention_39685497815574_kernel(
    const float* __restrict__ h,
    const unsigned char* __restrict__ adj,
    const float* __restrict__ w,
    const float* __restrict__ a_src,
    const float* __restrict__ a_dst,
    const float* __restrict__ bias,
    float* __restrict__ out)
{
    // LDS (58.3 KB): sStage (f32 h staging, TDM dest) overlays the later exp-attn sE
    __shared__ alignas(32) _Float16 sAP[NN * DIN];    // 16 KB: h f16 tile, reused as h_prime^T [o][node]
    __shared__ alignas(32) _Float16 sWt[DOUT * DIN];  //  8 KB: W^T  [o][i]
    __shared__ alignas(32) float    sStage[NN * DIN]; // 32 KB: h f32 (TDM), then sE f16 [i][j]
    __shared__ alignas(16) float sAs[DOUT], sAd[DOUT], sBias[DOUT];
    __shared__ alignas(16) float sSrc[NN], sDst[NN], sSum[NN];
    _Float16* sE = reinterpret_cast<_Float16*>(sStage);

    const int bid  = blockIdx.x;
    const int b    = bid >> 3;
    const int hh   = bid & 7;
    const int tid  = threadIdx.x;
    const int wave = tid >> 5;
    const int lane = tid & 31;
    const int l16  = lane & 15;
    const int lhi  = lane >> 4;
    const int m0   = wave * 16;

    // ---------------- stage h[b] into LDS ----------------
#if USE_TDM
    if (wave == 0) {
        // 2-D TDM descriptor: tile == tensor == 64 x 128 elements of 4 bytes
        unsigned long long ga = (unsigned long long)(uintptr_t)(h + (size_t)b * NN * DIN);
        unsigned int ldsa = (unsigned int)(uintptr_t)sStage;   // generic LDS ptr: low 32 = LDS byte addr
        v4u g0;
        g0[0] = 1u;                                            // count=1, no gather
        g0[1] = ldsa;                                          // lds_addr
        g0[2] = (unsigned int)ga;                              // global_addr[31:0]
        g0[3] = (unsigned int)((ga >> 32) & 0x1FFFFFFu) | 0x80000000u; // addr[56:32] | type=2
        v8i g1;
        g1[0] = 0x00020000;        // workgroup_mask=0, data_size=2 (4B)
        g1[1] = (DIN << 16);       // tensor_dim0 = 64  (bits 79:48)
        g1[2] = (NN  << 16);       // tensor_dim1 = 128 (bits 111:80)
        g1[3] = (DIN << 16);       // tile_dim0   = 64  (bits 127:112)
        g1[4] = NN;                // tile_dim1   = 128
        g1[5] = DIN;               // tensor_dim0_stride = 64
        g1[6] = 0;
        g1[7] = 0;
        v4i g2 = {0, 0, 0, 0};
        v4i g3 = {0, 0, 0, 0};
#if __clang_major__ >= 23
        v8i g4 = {0, 0, 0, 0, 0, 0, 0, 0};
        __builtin_amdgcn_tensor_load_to_lds(g0, g1, g2, g3, g4, 0);
#else
        __builtin_amdgcn_tensor_load_to_lds(g0, g1, g2, g3, 0);
#endif
        __builtin_amdgcn_s_wait_tensorcnt(0);
    }
#endif

    // ---------------- W^T (f16) + per-head vectors ----------------
    {
        const float* wp = w + (size_t)hh * DIN * DOUT;   // [i][o] row-major
        for (int i = tid; i < DIN * DOUT; i += 256) {
            int di = i >> 6, o = i & 63;
            sWt[o * DIN + di] = (_Float16)wp[i];
        }
        if (tid < DOUT) {
            sAs[tid]   = a_src[hh * DOUT + tid];
            sAd[tid]   = a_dst[hh * DOUT + tid];
            sBias[tid] = bias[tid];
        }
#if !USE_TDM
        const float4* hp4 = reinterpret_cast<const float4*>(h + (size_t)b * NN * DIN);
        for (int i = tid; i < NN * DIN / 4; i += 256) {
            float4 v = hp4[i];
            int o = i * 4;
            sAP[o + 0] = (_Float16)v.x; sAP[o + 1] = (_Float16)v.y;
            sAP[o + 2] = (_Float16)v.z; sAP[o + 3] = (_Float16)v.w;
        }
#endif
    }
    __syncthreads();

#if USE_TDM
    // convert staged f32 h tile -> f16 (LDS -> LDS)
    for (int i = tid; i < NN * DIN / 4; i += 256) {
        float4 v = reinterpret_cast<const float4*>(sStage)[i];
        int o = i * 4;
        sAP[o + 0] = (_Float16)v.x; sAP[o + 1] = (_Float16)v.y;
        sAP[o + 2] = (_Float16)v.z; sAP[o + 3] = (_Float16)v.w;
    }
    __syncthreads();
#endif

    // ---------------- GEMM1: h_prime = h @ W  (wave w -> rows 16w..16w+15) ----
    v8f acc[4] = {};
    v16h aF[2];
    #pragma unroll
    for (int kh = 0; kh < 2; ++kh) {
        const _Float16* p = sAP + (size_t)(m0 + l16) * DIN + kh * 32 + 16 * lhi;
        V16U u;
        u.h[0] = *reinterpret_cast<const v8h*>(p);
        u.h[1] = *reinterpret_cast<const v8h*>(p + 8);
        aF[kh] = u.v;
    }
    #pragma unroll
    for (int nb = 0; nb < 4; ++nb) {
        #pragma unroll
        for (int kh = 0; kh < 2; ++kh) {
            const _Float16* p = sWt + (size_t)(nb * 16 + l16) * DIN + kh * 32 + 16 * lhi;
            v16h bF = *reinterpret_cast<const v16h*>(p);
            acc[nb] = __builtin_amdgcn_wmma_f32_16x16x32_f16(
                false, aF[kh], false, bF, (short)0, acc[nb], false, false);
        }
    }
    __syncthreads();   // all GEMM1 LDS reads done before sAP is overwritten

    // ------ epilogue 1: h_prime^T (f16) to LDS; tanh; attn_src/dst reductions ----
    #pragma unroll
    for (int r = 0; r < 8; ++r) {
        int m = m0 + r + 8 * lhi;           // C/D layout: row = r + 8*(lane>>4)
        float sp = 0.f, dp = 0.f;
        #pragma unroll
        for (int nb = 0; nb < 4; ++nb) {
            float v = acc[nb][r];
            int o = nb * 16 + l16;
            sAP[o * NN + m] = (_Float16)v;  // transposed: GEMM2 B operand
            float t = fast_tanh(v);
            sp += t * sAs[o];
            dp += t * sAd[o];
        }
        #pragma unroll
        for (int off = 1; off < 16; off <<= 1) {
            sp += __shfl_xor(sp, off, 32);
            dp += __shfl_xor(dp, off, 32);
        }
        if (l16 == 0) { sSrc[m] = sp; sDst[m] = dp; }
    }
    __syncthreads();

    // ---------------- masked leaky-relu softmax (unnormalized exp into sE) -------
    // lane handles j = 4*lane .. 4*lane+3 -> one b32 adj load, b128 dst load, b64 sE store
    {
        const unsigned int* adj32 = reinterpret_cast<const unsigned int*>(adj + (size_t)b * NN * NN);
        float4 dv4 = reinterpret_cast<const float4*>(sDst)[lane];
        float dv[4] = {dv4.x, dv4.y, dv4.z, dv4.w};
        for (int ii = 0; ii < 16; ++ii) {
            int i = m0 + ii;
            float si = sSrc[i];
            unsigned int mbits = adj32[i * (NN / 4) + lane];
            float x[4]; bool mk[4];
            float mx = -3.0e30f;
            #pragma unroll
            for (int t4 = 0; t4 < 4; ++t4) {
                mk[t4] = ((mbits >> (8 * t4)) & 0xFFu) != 0u;
                float xx = si + dv[t4];
                xx = xx > 0.f ? xx : 0.2f * xx;            // leaky relu (0.2)
                x[t4] = xx;
                mx = fmaxf(mx, mk[t4] ? xx : -3.0e30f);
            }
            #pragma unroll
            for (int off = 1; off < 32; off <<= 1) mx = fmaxf(mx, __shfl_xor(mx, off, 32));
            union { _Float16 hv[4]; unsigned long long u64; } pk;
            float s = 0.f;
            #pragma unroll
            for (int t4 = 0; t4 < 4; ++t4) {
                float ev = mk[t4] ? __expf(x[t4] - mx) : 0.f;
                s += ev;
                pk.hv[t4] = (_Float16)ev;
            }
            *reinterpret_cast<unsigned long long*>(&sE[(size_t)i * NN + 4 * lane]) = pk.u64;
            #pragma unroll
            for (int off = 1; off < 32; off <<= 1) s += __shfl_xor(s, off, 32);
            if (lane == 0) sSum[i] = s;
        }
    }
    __syncthreads();

    // ---------------- GEMM2: out = E @ h_prime (K = 128) -------------------------
    v8f oac[4] = {};
    v16h eF[4];
    #pragma unroll
    for (int kh = 0; kh < 4; ++kh) {
        const _Float16* p = sE + (size_t)(m0 + l16) * NN + kh * 32 + 16 * lhi;
        V16U u;
        u.h[0] = *reinterpret_cast<const v8h*>(p);
        u.h[1] = *reinterpret_cast<const v8h*>(p + 8);
        eF[kh] = u.v;
    }
    #pragma unroll
    for (int nb = 0; nb < 4; ++nb) {
        #pragma unroll
        for (int kh = 0; kh < 4; ++kh) {
            const _Float16* p = sAP + (size_t)(nb * 16 + l16) * NN + kh * 32 + 16 * lhi;
            v16h bF = *reinterpret_cast<const v16h*>(p);
            oac[nb] = __builtin_amdgcn_wmma_f32_16x16x32_f16(
                false, eF[kh], false, bF, (short)0, oac[nb], false, false);
        }
    }

    // ---------------- epilogue: fold in 1/rowsum, bias; store --------------------
    float* outp = out + ((size_t)b * HEADS + hh) * NN * DOUT;
    #pragma unroll
    for (int r = 0; r < 8; ++r) {
        int i = m0 + r + 8 * lhi;
        float inv = 1.0f / sSum[i];
        #pragma unroll
        for (int nb = 0; nb < 4; ++nb) {
            int o = nb * 16 + l16;
            outp[(size_t)i * DOUT + o] = oac[nb][r] * inv + sBias[o];
        }
    }
}

extern "C" void kernel_launch(void* const* d_in, const int* in_sizes, int n_in,
                              void* d_out, int out_size, void* d_ws, size_t ws_size,
                              hipStream_t stream) {
    const float*         h     = (const float*)d_in[0];
    const unsigned char* adj   = (const unsigned char*)d_in[1];  // jax bool = 1 byte
    const float*         w     = (const float*)d_in[2];
    const float*         a_src = (const float*)d_in[3];
    const float*         a_dst = (const float*)d_in[4];
    const float*         bias  = (const float*)d_in[5];
    float*               out   = (float*)d_out;
    (void)in_sizes; (void)n_in; (void)out_size; (void)d_ws; (void)ws_size;

    dim3 grid(BSZ * HEADS);   // bid = b*8 + head -> consecutive blocks share h[b]/adj[b] in L2
    dim3 block(256);          // 8 wave32 waves
    MultiHeadGraphAttention_39685497815574_kernel<<<grid, block, 0, stream>>>(
        h, adj, w, a_src, a_dst, bias, out);
}
